// NormalityIndex_35235911696781
// MI455X (gfx1250) — matrix operand order repaired and checked by reading, not compiled, not run on hardware
//
#include <hip/hip_runtime.h>
#include <hip/hip_bf16.h>
#include <float.h>

typedef __attribute__((ext_vector_type(16))) __bf16 v16bf;
typedef __attribute__((ext_vector_type(8)))  __bf16 v8bf;
typedef __attribute__((ext_vector_type(8)))  float  v8f;

#define DIM     128
#define NQ      4096
#define NB      65536
#define KNN     9

#define MT      32            // queries per workgroup
#define NSPLIT  8             // N-dimension splits (merged in finalize)
#define NCHUNK  128           // bank rows staged per chunk
#define NPER    (NB / NSPLIT) // 8192 candidates per WG
#define CHUNKS  (NPER / NCHUNK)
#define ASTR    136           // LDS row stride (bf16 elems), pad vs 64-bank LDS
#define BSTR    136
#define DSTR    132           // f32 dist tile stride, pad so half-waves hit disjoint banks

// workspace layout (floats): [b_sq: NB][q_sq: NQ][partials: NQ*NSPLIT*KNN]
#define WS_BSQ  0
#define WS_QSQ  NB
#define WS_PART (NB + NQ)
// total ws use = (65536 + 4096 + 294912)*4 B ~= 1.4 MB

// ---------------- kernel 0: row sum-of-squares (f32 exact) ----------------
__global__ void knn_sumsq_kernel(const float* __restrict__ feat,
                                 const float* __restrict__ bank,
                                 float* __restrict__ ws) {
    int idx = blockIdx.x * blockDim.x + threadIdx.x;
    const float* src;
    float* dst;
    if (idx < NB)            { src = bank + (size_t)idx * DIM;        dst = ws + WS_BSQ + idx; }
    else if (idx < NB + NQ)  { src = feat + (size_t)(idx - NB) * DIM; dst = ws + WS_QSQ + (idx - NB); }
    else return;
    float s = 0.f;
    const float4* p = (const float4*)src;
#pragma unroll
    for (int i = 0; i < DIM / 4; ++i) {
        float4 v = p[i];
        s += v.x * v.x + v.y * v.y + v.z * v.z + v.w * v.w;
    }
    *dst = s;
}

__device__ __forceinline__ void bf16_split(float x, __bf16& hi, __bf16& lo) {
    hi = (__bf16)x;                 // RNE
    lo = (__bf16)(x - (float)hi);   // residual
}

__device__ __forceinline__ void topk_insert(float v, float* best) {
    if (v < best[KNN - 1]) {
        best[KNN - 1] = v;
#pragma unroll
        for (int j = KNN - 1; j > 0; --j) {   // single bubble pass restores order
            float lo = fminf(best[j - 1], best[j]);
            float hi = fmaxf(best[j - 1], best[j]);
            best[j - 1] = lo;
            best[j] = hi;
        }
    }
}

// ---------------- kernel 1: tiled WMMA distance + streaming top-9 ----------------
__launch_bounds__(256)
__global__ void knn_wmma_kernel(const float* __restrict__ feat,
                                const float* __restrict__ bank,
                                float* __restrict__ ws) {
    __shared__ __align__(16) __bf16 sAhi[MT * ASTR];
    __shared__ __align__(16) __bf16 sAlo[MT * ASTR];
    __shared__ __align__(16) __bf16 sBhi[NCHUNK * BSTR];
    __shared__ __align__(16) __bf16 sBlo[NCHUNK * BSTR];
    __shared__ __align__(16) float  sDist[MT * DSTR];

    const int tid  = threadIdx.x;
    const int mg   = blockIdx.x >> 3;     // query group
    const int sp   = blockIdx.x & 7;      // N split
    const int q0   = mg * MT;
    const int n0   = sp * NPER;

    // ---- stage A (queries) as bf16 hi/lo, once ----
    {
        int row = tid >> 3, seg = tid & 7;
        const float* src = feat + (size_t)(q0 + row) * DIM + seg * 16;
#pragma unroll
        for (int i = 0; i < 16; ++i) {
            __bf16 h, l;
            bf16_split(src[i], h, l);
            sAhi[row * ASTR + seg * 16 + i] = h;
            sAlo[row * ASTR + seg * 16 + i] = l;
        }
    }

    float best[KNN];
#pragma unroll
    for (int i = 0; i < KNN; ++i) best[i] = FLT_MAX;

    const int  lane = tid & 31;
    const int  wv   = tid >> 5;
    const bool lhi  = lane >= 16;
    const int  mrow = lane & 15;
    const float* bsq = ws + WS_BSQ;

    // Both tiles of a wave share the same mt -> hoist the wave's A fragments
    // into registers once (A is invariant across all chunks). Removes all A
    // LDS traffic from the steady-state loop (halves DS load count per tile).
    const int mt     = wv >> 2;            // 0..1
    const int ntbase = (wv & 3) * 2;       // {0,2,4,6}

    __syncthreads();                       // A staging visible
    v16bf aHi[4], aLo[4];
#pragma unroll
    for (int kk = 0; kk < 4; ++kk) {
        // A 16-bit 16x32 layout: lanes = M; dwords 0-3: K 0..7 (+8 hi-half),
        // dwords 4-7: K 16..23 (+8 hi-half)  -> two 16B loads per fragment
        int arow = mt * 16 + mrow;
        int acol = kk * 32 + (lhi ? 8 : 0);
        v8bf h0 = *(const v8bf*)&sAhi[arow * ASTR + acol];
        v8bf h1 = *(const v8bf*)&sAhi[arow * ASTR + acol + 16];
        v8bf l0 = *(const v8bf*)&sAlo[arow * ASTR + acol];
        v8bf l1 = *(const v8bf*)&sAlo[arow * ASTR + acol + 16];
#pragma unroll
        for (int i = 0; i < 8; ++i) {
            aHi[kk][i] = h0[i]; aHi[kk][8 + i] = h1[i];
            aLo[kk][i] = l0[i]; aLo[kk][8 + i] = l1[i];
        }
    }

    for (int c = 0; c < CHUNKS; ++c) {
        __syncthreads();   // sDist/sB reuse safe
        // ---- stage B chunk (128 bank rows) as bf16 hi/lo ----
        {
            int row = tid >> 1, half = tid & 1;
            const float* src = bank + (size_t)(n0 + c * NCHUNK + row) * DIM + half * 64;
            int col = half * 64;
#pragma unroll
            for (int i = 0; i < 64; ++i) {
                __bf16 h, l;
                bf16_split(src[i], h, l);
                sBhi[row * BSTR + col + i] = h;
                sBlo[row * BSTR + col + i] = l;
            }
        }
        __syncthreads();

        // ---- WMMA: each wave computes two 16x16 score tiles (same mt) ----
#pragma unroll
        for (int tt = 0; tt < 2; ++tt) {
            int nt = ntbase + tt;   // 0..7
            v8f acc = {};
#pragma unroll
            for (int kk = 0; kk < 4; ++kk) {
                // B 32x16 layout: lanes = N; elems 0-7: K 0..7 (+8 hi-half), elems 8-15: K 16..23 (+8 hi-half)
                int brow = nt * 16 + mrow;
                int bcol = kk * 32 + (lhi ? 8 : 0);
                v8bf bh0 = *(const v8bf*)&sBhi[brow * BSTR + bcol];
                v8bf bh1 = *(const v8bf*)&sBhi[brow * BSTR + bcol + 16];
                v8bf bl0 = *(const v8bf*)&sBlo[brow * BSTR + bcol];
                v8bf bl1 = *(const v8bf*)&sBlo[brow * BSTR + bcol + 16];
                v16bf bhi, blo;
#pragma unroll
                for (int i = 0; i < 8; ++i) {
                    bhi[i] = bh0[i]; bhi[8 + i] = bh1[i];
                    blo[i] = bl0[i]; blo[8 + i] = bl1[i];
                }
                // bf16x3: hi*hi + hi*lo + lo*hi  ~= f32 dot
                acc = __builtin_amdgcn_wmma_f32_16x16x32_bf16(false, aHi[kk], false, bhi, (short)0, acc, false, false);
                acc = __builtin_amdgcn_wmma_f32_16x16x32_bf16(false, aHi[kk], false, blo, (short)0, acc, false, false);
                acc = __builtin_amdgcn_wmma_f32_16x16x32_bf16(false, aLo[kk], false, bhi, (short)0, acc, false, false);
            }
            // D layout: lanes0-15 -> n=lane, m=r ; lanes16-31 -> n=lane-16, m=r+8
            int ncl = nt * 16 + (lane & 15);
            float bs = bsq[n0 + c * NCHUNK + ncl];
            int mbase = mt * 16 + (lhi ? 8 : 0);
#pragma unroll
            for (int r = 0; r < 8; ++r) {
                sDist[(mbase + r) * DSTR + ncl] = bs - 2.0f * acc[r];   // q_sq added later
            }
        }
        __syncthreads();

        // ---- streaming top-9: 8 threads per query scan the 128 chunk scores ----
        {
            int q = tid >> 3, s = tid & 7;
#pragma unroll
            for (int i = 0; i < NCHUNK / 8; ++i) {
                topk_insert(sDist[q * DSTR + s + i * 8], best);
            }
        }
    }

    // ---- merge the 8 per-thread lists of each query inside this split ----
    __syncthreads();
    {
        int q = tid >> 3, s = tid & 7;
#pragma unroll
        for (int i = 0; i < KNN; ++i) sDist[q * (8 * KNN) + s * KNN + i] = best[i];
    }
    __syncthreads();
    if (tid < MT) {
        float b2[KNN];
#pragma unroll
        for (int i = 0; i < KNN; ++i) b2[i] = FLT_MAX;
        for (int i = 0; i < 8 * KNN; ++i) topk_insert(sDist[tid * (8 * KNN) + i], b2);
        float* part = ws + WS_PART + ((size_t)(q0 + tid) * NSPLIT + sp) * KNN;
#pragma unroll
        for (int i = 0; i < KNN; ++i) part[i] = b2[i];
    }
}

// ---------------- kernel 2: merge splits, sqrt, mean ----------------
__global__ void knn_finalize_kernel(const float* __restrict__ ws, float* __restrict__ out) {
    int q = blockIdx.x * blockDim.x + threadIdx.x;
    if (q >= NQ) return;
    const float* part = ws + WS_PART + (size_t)q * NSPLIT * KNN;
    float best[KNN];
#pragma unroll
    for (int i = 0; i < KNN; ++i) best[i] = FLT_MAX;
    for (int i = 0; i < NSPLIT * KNN; ++i) topk_insert(part[i], best);
    float qsq = ws[WS_QSQ + q];
    float s = 0.f;
#pragma unroll
    for (int i = 0; i < KNN; ++i) s += sqrtf(fmaxf(qsq + best[i], 0.f));
    out[q] = s * (1.0f / KNN);
}

extern "C" void kernel_launch(void* const* d_in, const int* in_sizes, int n_in,
                              void* d_out, int out_size, void* d_ws, size_t ws_size,
                              hipStream_t stream) {
    const float* feat = (const float*)d_in[0];   // [4096,128]
    const float* bank = (const float*)d_in[1];   // [65536,128]
    float* out = (float*)d_out;                  // [4096]
    float* ws  = (float*)d_ws;                   // needs ~1.4 MB

    knn_sumsq_kernel<<<(NB + NQ + 255) / 256, 256, 0, stream>>>(feat, bank, ws);
    knn_wmma_kernel<<<(NQ / MT) * NSPLIT, 256, 0, stream>>>(feat, bank, ws);
    knn_finalize_kernel<<<(NQ + 255) / 256, 256, 0, stream>>>(ws, out);
}
